// FSADAttention3D_64201171141224
// MI455X (gfx1250) — matrix operand order repaired and checked by reading, not compiled, not run on hardware
//
#include <hip/hip_runtime.h>
#include <hip/hip_bf16.h>
#include <stdint.h>

// ---------------- problem dims ----------------
#define B_   2
#define QC_  128
#define DD   16
#define HH   32
#define WW   32
#define NVOX (DD*HH*WW)      // 16384 voxels per batch
#define HID_ 128
#define NH_  4
#define NP__ 4
#define NS_  3
#define HD_  32              // HID / NH
#define MRAW 192             // 144 offset channels + 48 attn-weight channels
#define OFF_SCALE 0.35f
#define EPS_ 1e-5f

typedef __attribute__((ext_vector_type(16))) __bf16 v16bf;
typedef __attribute__((ext_vector_type(8)))  float  v8f;

__device__ __forceinline__ __bf16 to_bf16(float f) {
  unsigned u = __builtin_bit_cast(unsigned, f);
  unsigned r = u + 0x7FFFu + ((u >> 16) & 1u);   // round-to-nearest-even
  return __builtin_bit_cast(__bf16, (unsigned short)(r >> 16));
}
__device__ __forceinline__ int imin(int a, int b) { return a < b ? a : b; }

// ---------------------------------------------------------------------------
// K0: Wc[192,128] = [ow;ww] (192x128, HID cols) x qw (128x128) ; biasc = [ob;wb]
// ---------------------------------------------------------------------------
__global__ __launch_bounds__(256)
void combine_w_kernel(const float* __restrict__ qw,
                      const float* __restrict__ ow, const float* __restrict__ ob,
                      const float* __restrict__ ww, const float* __restrict__ wb,
                      float* __restrict__ Wc, float* __restrict__ biasc)
{
  int idx = blockIdx.x * blockDim.x + threadIdx.x;
  if (idx < MRAW * QC_) {
    int r = idx >> 7;           // output row (0..191)
    int k = idx & 127;          // QC column
    const float* src = (r < 144) ? (ow + (size_t)r * HID_) : (ww + (size_t)(r - 144) * HID_);
    float s = 0.f;
    for (int j = 0; j < HID_; ++j) s = fmaf(src[j], qw[(size_t)j * QC_ + k], s);
    Wc[idx] = s;
  }
  if (idx < MRAW) biasc[idx] = (idx < 144) ? ob[idx] : wb[idx - 144];
}

// ---------------------------------------------------------------------------
// WMMA GEMM: Dst[b] (MxN) = A (MxK, f32 row-major) * X[b] (KxN) (+ bias[m])
// One wave owns a 16(M) x 64(N) strip: 4 accumulators share one A fragment.
// A panel (16 x K f32) is staged to LDS via global_load_async_to_lds_b128,
// then the fully-unrolled K loop (K compile-time) issues 4 wmma per K-step.
// CDNA5 wave32 fragment layouts per ISA 7.12.2.
// ---------------------------------------------------------------------------
template<int K, bool HAS_BIAS>
__global__ __launch_bounds__(256)
void wmma_gemm_kernel(const float* __restrict__ A,
                      const float* __restrict__ X,
                      const float* __restrict__ bias,
                      float* __restrict__ Dst,
                      int M, int N,
                      long strideX, long strideD, int totalWaves)
{
  constexpr int NT = 4;                       // N tiles per wave (64 columns)
  __shared__ float shA[8][16 * K];            // 8 waves x 8KB = 64KB

  int waveId = (int)((blockIdx.x * blockDim.x + threadIdx.x) >> 5);
  if (waveId >= totalWaves) return;
  int waveInBlk = threadIdx.x >> 5;
  int lane = threadIdx.x & 31;
  int ml = lane & 15;          // row (A) / col (B,D) within tile
  int h  = lane >> 4;          // lane half

  int tilesN = N >> 6;                        // strips of 64
  int tilesM = M >> 4;
  int tilesPerBatch = tilesM * tilesN;
  int b  = waveId / tilesPerBatch;
  int t  = waveId % tilesPerBatch;
  int m0 = (t / tilesN) << 4;
  int n0 = (t % tilesN) << 6;

  const float* Xb = X + (size_t)b * strideX;
  float*       Db = Dst + (size_t)b * strideD;

  // ---- async stage of this wave's A panel (16 rows x K f32) into LDS ----
  {
    const float* Apan = A + (size_t)m0 * K;
    uint32_t ldsbase = (uint32_t)(uintptr_t)&shA[waveInBlk][0];
    // 16*K dwords, 32 lanes x 4 dwords per op
    for (int i = 0; i < (16 * K) / (32 * 4); ++i) {
      uint32_t ldso = ldsbase + (uint32_t)((i * 32 + lane) * 16);
      uint64_t ga   = (uint64_t)(uintptr_t)(Apan + (size_t)(i * 32 + lane) * 4);
      asm volatile("global_load_async_to_lds_b128 %0, %1, off"
                   :: "v"(ldso), "v"(ga) : "memory");
    }
    asm volatile("s_wait_asynccnt 0x0" ::: "memory");
  }
  const float* Alds = &shA[waveInBlk][0];     // [16][K]

  v8f c0 = {}, c1 = {}, c2 = {}, c3 = {};
  const float* Xc0 = Xb + (n0 + 0  + ml);
  const float* Xc1 = Xb + (n0 + 16 + ml);
  const float* Xc2 = Xb + (n0 + 32 + ml);
  const float* Xc3 = Xb + (n0 + 48 + ml);

  #pragma unroll
  for (int k0 = 0; k0 < K; k0 += 32) {
    if (k0 + 32 < K) {          // compile-time after unroll: prefetch next panels
      __builtin_prefetch(Xc0 + (size_t)(k0 + 32) * N, 0, 3);
      __builtin_prefetch(Xc2 + (size_t)(k0 + 32) * N, 0, 3);
    }
    v16bf a;
    #pragma unroll
    for (int e = 0; e < 16; ++e) {
      // A 16x32 bf16: e<8 -> K=e+8h ; e>=8 -> K=16+(e-8)+8h
      int ka = (e < 8) ? (e + h * 8) : (e + 8 + h * 8);
      a[e] = to_bf16(Alds[(size_t)ml * K + k0 + ka]);
    }
    v16bf b0, b1, b2, b3;
    #pragma unroll
    for (int e = 0; e < 16; ++e) {
      // B 32x16 bf16: e<8 -> K=e+8h ; e>=8 -> K=16+(e-8)+8h
      size_t kro = (size_t)(k0 + ((e < 8) ? (e + h * 8) : (e + 8 + h * 8))) * N;
      b0[e] = to_bf16(Xc0[kro]);
      b1[e] = to_bf16(Xc1[kro]);
      b2[e] = to_bf16(Xc2[kro]);
      b3[e] = to_bf16(Xc3[kro]);
    }
    c0 = __builtin_amdgcn_wmma_f32_16x16x32_bf16(false, a, false, b0, (short)0, c0, false, false);
    c1 = __builtin_amdgcn_wmma_f32_16x16x32_bf16(false, a, false, b1, (short)0, c1, false, false);
    c2 = __builtin_amdgcn_wmma_f32_16x16x32_bf16(false, a, false, b2, (short)0, c2, false, false);
    c3 = __builtin_amdgcn_wmma_f32_16x16x32_bf16(false, a, false, b3, (short)0, c3, false, false);
  }

  #pragma unroll
  for (int r = 0; r < 8; ++r) {
    int m = m0 + r + h * 8;    // D layout: vgpr r, lane-half h -> M = r + 8h
    float bv = HAS_BIAS ? bias[m] : 0.f;
    float* drow = Db + (size_t)m * N + n0 + ml;
    drow[0]  = c0[r] + bv;
    drow[16] = c1[r] + bv;
    drow[32] = c2[r] + bv;
    drow[48] = c3[r] + bv;
  }
}

// ---------------------------------------------------------------------------
// K2: in-place tanh*0.35 on first 144 channels; softmax over 12 per head on last 48
// raw layout: [B, 192, NVOX]
// ---------------------------------------------------------------------------
__global__ __launch_bounds__(256)
void postproc_kernel(float* __restrict__ raw)
{
  int idx = blockIdx.x * blockDim.x + threadIdx.x;   // over B*NVOX
  if (idx >= B_ * NVOX) return;
  int b = idx / NVOX, n = idx % NVOX;
  float* base = raw + (size_t)b * MRAW * NVOX + n;

  for (int ch = 0; ch < 144; ++ch) {
    float v = base[(size_t)ch * NVOX];
    base[(size_t)ch * NVOX] = tanhf(v) * OFF_SCALE;
  }
  for (int hd = 0; hd < NH_; ++hd) {
    float* p = base + (size_t)(144 + hd * 12) * NVOX;
    float vals[12];
    float mx = -3.4e38f;
    #pragma unroll
    for (int i = 0; i < 12; ++i) { vals[i] = p[(size_t)i * NVOX]; mx = fmaxf(mx, vals[i]); }
    float sum = 0.f;
    #pragma unroll
    for (int i = 0; i < 12; ++i) { vals[i] = __expf(vals[i] - mx); sum += vals[i]; }
    float inv = 1.f / sum;
    #pragma unroll
    for (int i = 0; i < 12; ++i) p[(size_t)i * NVOX] = vals[i] * inv;
  }
}

// ---------------------------------------------------------------------------
// K3: fused trilinear sampling of RAW values + on-the-fly head projection.
// Linearity: proj(sample(v)) == sample(proj(v)), so we avoid materializing the
// 134 MB projected pyramid. acc layout: [B, NH*HD=128, NVOX].
// ---------------------------------------------------------------------------
template<int Cs, int Dv, int Hv, int Wv>
__device__ __forceinline__ void sample_scale(
    const float* __restrict__ val,   // this batch: [Cs, Dv, Hv, Wv]
    const float* __restrict__ vw,    // rows for this head: [HD_, Cs]
    const float* __restrict__ offb,  // &raw[b, 0, n]
    int head, int s, float bx, float by, float bz,
    float* __restrict__ accv)
{
  const int cstride = Dv * Hv * Wv;
  #pragma unroll
  for (int p = 0; p < NP__; ++p) {
    const int chb = ((head * NS_ + s) * NP__ + p) * 3;
    float ox = offb[(size_t)(chb + 0) * NVOX];
    float oy = offb[(size_t)(chb + 1) * NVOX];
    float oz = offb[(size_t)(chb + 2) * NVOX];
    float aw = offb[(size_t)(144 + head * 12 + s * NP__ + p) * NVOX];

    // align_corners=False unnormalize + border clamp
    float x = fminf(fmaxf(((bx + ox + 1.f) * Wv - 1.f) * 0.5f, 0.f), (float)(Wv - 1));
    float y = fminf(fmaxf(((by + oy + 1.f) * Hv - 1.f) * 0.5f, 0.f), (float)(Hv - 1));
    float z = fminf(fmaxf(((bz + oz + 1.f) * Dv - 1.f) * 0.5f, 0.f), (float)(Dv - 1));
    float xf = floorf(x), yf = floorf(y), zf = floorf(z);
    int x0 = (int)xf, y0 = (int)yf, z0 = (int)zf;
    int x1 = imin(x0 + 1, Wv - 1), y1 = imin(y0 + 1, Hv - 1), z1 = imin(z0 + 1, Dv - 1);
    float wx = x - xf, wy = y - yf, wz = z - zf;

    float w000 = (1.f-wx)*(1.f-wy)*(1.f-wz)*aw, w001 = wx*(1.f-wy)*(1.f-wz)*aw;
    float w010 = (1.f-wx)*wy*(1.f-wz)*aw,       w011 = wx*wy*(1.f-wz)*aw;
    float w100 = (1.f-wx)*(1.f-wy)*wz*aw,       w101 = wx*(1.f-wy)*wz*aw;
    float w110 = (1.f-wx)*wy*wz*aw,             w111 = wx*wy*wz*aw;

    int i000 = (z0*Hv + y0)*Wv + x0, i001 = (z0*Hv + y0)*Wv + x1;
    int i010 = (z0*Hv + y1)*Wv + x0, i011 = (z0*Hv + y1)*Wv + x1;
    int i100 = (z1*Hv + y0)*Wv + x0, i101 = (z1*Hv + y0)*Wv + x1;
    int i110 = (z1*Hv + y1)*Wv + x0, i111 = (z1*Hv + y1)*Wv + x1;

    #pragma unroll 1
    for (int c = 0; c < Cs; ++c) {
      const float* vc = val + (size_t)c * cstride;
      float samp = w000*vc[i000] + w001*vc[i001] + w010*vc[i010] + w011*vc[i011]
                 + w100*vc[i100] + w101*vc[i101] + w110*vc[i110] + w111*vc[i111];
      const float* wcol = vw + c;
      #pragma unroll
      for (int hd = 0; hd < HD_; ++hd)
        accv[hd] = fmaf(samp, wcol[(size_t)hd * Cs], accv[hd]);
    }
  }
}

__global__ __launch_bounds__(256)
void sample_project_kernel(const float* __restrict__ raw,
                           const float* __restrict__ v0, const float* __restrict__ v1,
                           const float* __restrict__ v2,
                           const float* __restrict__ vw0, const float* __restrict__ vw1,
                           const float* __restrict__ vw2,
                           float* __restrict__ acc)
{
  int bh = blockIdx.y;          // B*NH = 8
  int b = bh >> 2, head = bh & 3;
  int n = blockIdx.x * blockDim.x + threadIdx.x;   // 0..NVOX-1
  int wi = n & (WW - 1), hi = (n / WW) & (HH - 1), di = n / (WW * HH);
  float bx = -1.f + 2.f * wi / (float)(WW - 1);
  float by = -1.f + 2.f * hi / (float)(HH - 1);
  float bz = -1.f + 2.f * di / (float)(DD - 1);
  const float* offb = raw + (size_t)b * MRAW * NVOX + n;

  float accv[HD_];
  #pragma unroll
  for (int i = 0; i < HD_; ++i) accv[i] = 0.f;

  sample_scale< 32, 32, 64, 64>(v0 + (size_t)b * 32 * 32 * 64 * 64,
                                vw0 + (size_t)head * HD_ * 32,  offb, head, 0, bx, by, bz, accv);
  sample_scale< 64, 16, 32, 32>(v1 + (size_t)b * 64 * 16 * 32 * 32,
                                vw1 + (size_t)head * HD_ * 64,  offb, head, 1, bx, by, bz, accv);
  sample_scale<128,  8, 16, 16>(v2 + (size_t)b * 128 * 8 * 16 * 16,
                                vw2 + (size_t)head * HD_ * 128, offb, head, 2, bx, by, bz, accv);

  float* ap = acc + ((size_t)b * HID_ + head * HD_) * NVOX + n;
  #pragma unroll
  for (int hd = 0; hd < HD_; ++hd) ap[(size_t)hd * NVOX] = accv[hd];
}

// ---------------------------------------------------------------------------
// K5: per-(b,c) mean / inv_std over NVOX (biased variance, InstanceNorm)
// ---------------------------------------------------------------------------
__global__ __launch_bounds__(256)
void stats_kernel(const float* __restrict__ pre, float* __restrict__ stats)
{
  int bc = blockIdx.x;                         // 0 .. B*128-1
  const float* p = pre + (size_t)bc * NVOX;
  float s = 0.f, ss = 0.f;
  for (int i = threadIdx.x; i < NVOX; i += blockDim.x) {
    float v = p[i]; s += v; ss = fmaf(v, v, ss);
  }
  __shared__ float sh[256], sh2[256];
  sh[threadIdx.x] = s; sh2[threadIdx.x] = ss;
  __syncthreads();
  for (int st = 128; st > 0; st >>= 1) {
    if ((int)threadIdx.x < st) { sh[threadIdx.x] += sh[threadIdx.x + st]; sh2[threadIdx.x] += sh2[threadIdx.x + st]; }
    __syncthreads();
  }
  if (threadIdx.x == 0) {
    float m = sh[0] * (1.f / NVOX);
    float var = sh2[0] * (1.f / NVOX) - m * m;
    stats[bc * 2 + 0] = m;
    stats[bc * 2 + 1] = rsqrtf(var + EPS_);
  }
}

// ---------------------------------------------------------------------------
// K6: apply InstanceNorm affine
// ---------------------------------------------------------------------------
__global__ __launch_bounds__(256)
void norm_kernel(const float* __restrict__ pre, const float* __restrict__ stats,
                 const float* __restrict__ gamma, const float* __restrict__ beta,
                 float* __restrict__ out)
{
  long idx = (long)blockIdx.x * blockDim.x + threadIdx.x;
  if (idx >= (long)B_ * HID_ * NVOX) return;
  int bc = (int)(idx / NVOX);
  int c = bc & 127;
  float m = stats[bc * 2 + 0], is = stats[bc * 2 + 1];
  out[idx] = (pre[idx] - m) * is * gamma[c] + beta[c];
}

// ---------------------------------------------------------------------------
extern "C" void kernel_launch(void* const* d_in, const int* in_sizes, int n_in,
                              void* d_out, int out_size, void* d_ws, size_t ws_size,
                              hipStream_t stream)
{
  const float* query = (const float*)d_in[0];
  const float* v0    = (const float*)d_in[1];
  const float* v1    = (const float*)d_in[2];
  const float* v2    = (const float*)d_in[3];
  const float* qw    = (const float*)d_in[4];
  const float* vw0   = (const float*)d_in[5];
  const float* vw1   = (const float*)d_in[6];
  const float* vw2   = (const float*)d_in[7];
  const float* ow    = (const float*)d_in[8];
  const float* ob    = (const float*)d_in[9];
  const float* ww    = (const float*)d_in[10];
  const float* wb    = (const float*)d_in[11];
  const float* outw  = (const float*)d_in[12];
  const float* gamma = (const float*)d_in[13];
  const float* beta  = (const float*)d_in[14];
  float* out = (float*)d_out;

  // workspace carve-out (all buffers fully overwritten every call)
  char* ws = (char*)d_ws;
  size_t cur = 0;
  auto alloc = [&](size_t nfloats) {
    float* p = (float*)(ws + cur);
    cur += ((nfloats * sizeof(float) + 255) / 256) * 256;
    return p;
  };
  float* Wc    = alloc((size_t)MRAW * QC_);          //  98 KB combined head weight
  float* biasc = alloc(MRAW);
  float* raw   = alloc((size_t)B_ * MRAW * NVOX);    //  25 MB off+aw
  float* acc   = alloc((size_t)B_ * HID_ * NVOX);    //  17 MB sampled+projected
  float* pre   = alloc((size_t)B_ * HID_ * NVOX);    //  17 MB pre-norm output
  float* stats = alloc((size_t)B_ * HID_ * 2);
  (void)ws_size; (void)in_sizes; (void)n_in; (void)out_size;

  // K0: fold qw into offset/weight heads: Wc = [ow;ww] * qw
  combine_w_kernel<<<(MRAW * QC_ + 255) / 256, 256, 0, stream>>>(qw, ow, ob, ww, wb, Wc, biasc);

  // K1: raw[b,0:192,n] = Wc * query_feature (WMMA bf16, +bias)
  {
    int M = MRAW, N = NVOX;
    int totalWaves = B_ * (M / 16) * (N / 64);
    int blocks = (totalWaves * 32 + 255) / 256;
    wmma_gemm_kernel<QC_, true><<<blocks, 256, 0, stream>>>(
        Wc, query, biasc, raw, M, N,
        (long)QC_ * NVOX, (long)MRAW * NVOX, totalWaves);
  }

  // K2: tanh*scale on offsets, softmax(12) on attention weights (in place)
  postproc_kernel<<<(B_ * NVOX + 255) / 256, 256, 0, stream>>>(raw);

  // K3: fused trilinear sample of raw values + head projection -> acc
  {
    dim3 grid(NVOX / 256, B_ * NH_);
    sample_project_kernel<<<grid, 256, 0, stream>>>(raw, v0, v1, v2, vw0, vw1, vw2, acc);
  }

  // K4: pre = outw * acc (WMMA bf16, no bias)
  {
    int M = HID_, N = NVOX;
    int totalWaves = B_ * (M / 16) * (N / 64);
    int blocks = (totalWaves * 32 + 255) / 256;
    wmma_gemm_kernel<HID_, false><<<blocks, 256, 0, stream>>>(
        outw, acc, nullptr, pre, M, N,
        (long)HID_ * NVOX, (long)HID_ * NVOX, totalWaves);
  }

  // K5 + K6: InstanceNorm3d (biased var) + affine -> d_out
  stats_kernel<<<B_ * HID_, 256, 0, stream>>>(pre, stats);
  norm_kernel<<<(int)(((long)B_ * HID_ * NVOX + 255) / 256), 256, 0, stream>>>(pre, stats, gamma, beta, out);
}
